// Olmo2AHAAttention_1159641170239
// MI455X (gfx1250) — matrix operand order repaired and checked by reading, not compiled, not run on hardware
//
#include <hip/hip_runtime.h>
#include <hip/hip_bf16.h>

// ---------------- constants ----------------
#define T_LEN 2048
#define HID   2048
#define NH    16
#define HD    128
#define NQG   2064   // H*D + H
#define WIN   128

typedef __attribute__((ext_vector_type(16))) __bf16 v16bf;
typedef __attribute__((ext_vector_type(8)))  __bf16 v8bf;
typedef __attribute__((ext_vector_type(8)))  float  v8f;
typedef __attribute__((ext_vector_type(8)))  unsigned short v8u16;

union Frag { v16bf v; v8bf h[2]; };

__device__ __forceinline__ unsigned short f2b(float f) {
    union { float f; unsigned int u; } x;
    x.f = f;
    unsigned int r = x.u + 0x7FFFu + ((x.u >> 16) & 1u);
    return (unsigned short)(r >> 16);
}

__device__ __forceinline__ v8f wmma_bf16(v16bf a, v16bf b, v8f c) {
    return __builtin_amdgcn_wmma_f32_16x16x32_bf16(false, a, false, b, (short)0, c, false, false);
}

__device__ __forceinline__ void lds_fence() {
    __builtin_amdgcn_wave_barrier();
    asm volatile("s_wait_dscnt 0" ::: "memory");
}

// ---------------- generic f32-in / f32-out GEMM via bf16 WMMA ----------------
// C[M,N] = A[M,K] * B[K,N], all row-major f32; converted to bf16 in LDS.
#define GBM 128
#define GBN 128
#define GBK 32

__global__ __launch_bounds__(256) void gemm_bf16_kernel(
    const float* __restrict__ A, const float* __restrict__ B, float* __restrict__ C,
    int M, int N, int K)
{
    __shared__ __align__(16) unsigned short As[GBM * GBK];   // row-major [m][k]
    __shared__ __align__(16) unsigned short Bs[GBN * GBK];   // col-major [n][k]
    unsigned int* Bs32 = (unsigned int*)Bs;

    const int tid  = threadIdx.x;
    const int lane = tid & 31;
    const int wave = tid >> 5;
    const int hi   = lane >> 4;
    const int lx   = lane & 15;

    const int m0 = blockIdx.y * GBM;
    const int n0 = blockIdx.x * GBN;
    const int wm = (wave >> 1) * 32;   // 0,32,64,96
    const int wn = (wave & 1) * 64;    // 0,64

    // block-uniform fast path: tile fully in-range (true for all tiles except
    // the single partial N-tile of the Wq projection)
    const bool full = (m0 + GBM <= M) && (n0 + GBN <= N) && ((K & (GBK - 1)) == 0);

    v8f acc[2][4];
#pragma unroll
    for (int i = 0; i < 2; i++)
#pragma unroll
        for (int j = 0; j < 4; j++)
#pragma unroll
            for (int r = 0; r < 8; r++) acc[i][j][r] = 0.0f;

    for (int k0 = 0; k0 < K; k0 += GBK) {
        if (full) {
            // ---- fast path: b128 loads, packed LDS stores ----
            {   // A tile: 128 rows x 32 k; thread -> (row, 16 consecutive k)
                const int row = tid >> 1;
                const int kp  = (tid & 1) * 16;
                const float* src = A + (size_t)(m0 + row) * K + k0 + kp;
                __builtin_prefetch(src + GBK, 0, 3);
                const float4 f0 = *(const float4*)(src);
                const float4 f1 = *(const float4*)(src + 4);
                const float4 f2 = *(const float4*)(src + 8);
                const float4 f3 = *(const float4*)(src + 12);
                v8u16 p0, p1;
                p0[0] = f2b(f0.x); p0[1] = f2b(f0.y); p0[2] = f2b(f0.z); p0[3] = f2b(f0.w);
                p0[4] = f2b(f1.x); p0[5] = f2b(f1.y); p0[6] = f2b(f1.z); p0[7] = f2b(f1.w);
                p1[0] = f2b(f2.x); p1[1] = f2b(f2.y); p1[2] = f2b(f2.z); p1[3] = f2b(f2.w);
                p1[4] = f2b(f3.x); p1[5] = f2b(f3.y); p1[6] = f2b(f3.z); p1[7] = f2b(f3.w);
                *(v8u16*)&As[row * GBK + kp]     = p0;
                *(v8u16*)&As[row * GBK + kp + 8] = p1;
            }
            {   // B tile: 32 k x 128 n; thread -> (2 k-rows, 8 consecutive n)
                const int kr = (tid & 15) * 2;
                const int nc = (tid >> 4) * 8;
                const float* s0 = B + (size_t)(k0 + kr) * N + n0 + nc;
                const float* s1 = s0 + N;
                __builtin_prefetch(s0 + (size_t)GBK * N, 0, 3);
                const float4 a0 = *(const float4*)(s0);
                const float4 a1 = *(const float4*)(s0 + 4);
                const float4 b0 = *(const float4*)(s1);
                const float4 b1 = *(const float4*)(s1 + 4);
                const float r0[8] = {a0.x, a0.y, a0.z, a0.w, a1.x, a1.y, a1.z, a1.w};
                const float r1[8] = {b0.x, b0.y, b0.z, b0.w, b1.x, b1.y, b1.z, b1.w};
#pragma unroll
                for (int i = 0; i < 8; i++) {
                    const unsigned int pk =
                        (unsigned int)f2b(r0[i]) | ((unsigned int)f2b(r1[i]) << 16);
                    Bs32[((nc + i) * GBK + kr) >> 1] = pk;
                }
            }
        } else {
            // ---- slow path: per-element bounds-checked (edge tiles only) ----
            {
                const int row = tid >> 1;
                const int kp  = (tid & 1) * 16;
                const int gm  = m0 + row;
                const float* src = A + (size_t)gm * K + k0 + kp;
#pragma unroll
                for (int i = 0; i < 16; i++) {
                    float v = 0.0f;
                    const int gk = k0 + kp + i;
                    if (gm < M && gk < K) v = src[i];
                    As[row * GBK + kp + i] = f2b(v);
                }
            }
            {
                const int kr = tid >> 3;          // 0..31
                const int cp = (tid & 7) * 16;    // 0..112
                const int gk = k0 + kr;
                const float* src = B + (size_t)gk * N + n0 + cp;
#pragma unroll
                for (int i = 0; i < 16; i++) {
                    float v = 0.0f;
                    const int gn = n0 + cp + i;
                    if (gk < K && gn < N) v = src[i];
                    Bs[(cp + i) * GBK + kr] = f2b(v);
                }
            }
        }
        __syncthreads();

        Frag af[2], bf[4];
#pragma unroll
        for (int i = 0; i < 2; i++) {
            const unsigned short* p = &As[(wm + i * 16 + lx) * GBK];
            af[i].h[0] = *(const v8bf*)(p + 8 * hi);
            af[i].h[1] = *(const v8bf*)(p + 16 + 8 * hi);
        }
#pragma unroll
        for (int j = 0; j < 4; j++) {
            const unsigned short* p = &Bs[(wn + j * 16 + lx) * GBK + 8 * hi];
            bf[j].h[0] = *(const v8bf*)(p);
            bf[j].h[1] = *(const v8bf*)(p + 16);
        }
#pragma unroll
        for (int i = 0; i < 2; i++)
#pragma unroll
            for (int j = 0; j < 4; j++)
                acc[i][j] = wmma_bf16(af[i].v, bf[j].v, acc[i][j]);
        __syncthreads();
    }

#pragma unroll
    for (int i = 0; i < 2; i++)
#pragma unroll
        for (int j = 0; j < 4; j++)
#pragma unroll
            for (int r = 0; r < 8; r++) {
                const int gm = m0 + wm + i * 16 + r + 8 * hi;
                const int gn = n0 + wn + j * 16 + lx;
                if (gm < M && gn < N) C[(size_t)gm * N + gn] = acc[i][j][r];
            }
}

// ---------------- RMSNorm + RoPE for Q and K, f32 -> bf16 [h][t][d] ----------------
__global__ __launch_bounds__(256) void norm_rope_kernel(
    const float* __restrict__ qg, const float* __restrict__ kf,
    const float* __restrict__ qw, const float* __restrict__ kw,
    unsigned short* __restrict__ qbf, unsigned short* __restrict__ kbf)
{
    const int t   = blockIdx.x;
    const int tid = threadIdx.x;
    __shared__ float red[256];
    __shared__ float stats[2];

    const float* qrow = qg + (size_t)t * NQG;
    const float* krow = kf + (size_t)t * HID;

    float sq = 0.0f, sk = 0.0f;
    for (int i = tid; i < HID; i += 256) {
        float a = qrow[i]; sq += a * a;
        float b = krow[i]; sk += b * b;
    }
    red[tid] = sq; __syncthreads();
    for (int s = 128; s > 0; s >>= 1) { if (tid < s) red[tid] += red[tid + s]; __syncthreads(); }
    if (tid == 0) stats[0] = rsqrtf(red[0] * (1.0f / HID) + 1e-6f);
    __syncthreads();
    red[tid] = sk; __syncthreads();
    for (int s = 128; s > 0; s >>= 1) { if (tid < s) red[tid] += red[tid + s]; __syncthreads(); }
    if (tid == 0) stats[1] = rsqrtf(red[0] * (1.0f / HID) + 1e-6f);
    __syncthreads();

    const float rq = stats[0];
    const float rk = stats[1];

    for (int i = tid; i < HID; i += 256) {
        const int h = i >> 7;
        const int d = i & 127;
        const int j = d & 63;
        const float inv = powf(500000.0f, -(float)(2 * j) / (float)HD);
        const float ang = (float)t * inv;
        const float c = cosf(ang), s = sinf(ang);
        const int pidx = h * HD + ((d < 64) ? (d + 64) : (d - 64));

        float xd = qrow[i] * rq * qw[i];
        float xp = qrow[pidx] * rq * qw[pidx];
        float oq = (d < 64) ? (xd * c - xp * s) : (xd * c + xp * s);
        qbf[(size_t)h * T_LEN * HD + (size_t)t * HD + d] = f2b(oq);

        float yd = krow[i] * rk * kw[i];
        float yp = krow[pidx] * rk * kw[pidx];
        float ok = (d < 64) ? (yd * c - yp * s) : (yd * c + yp * s);
        kbf[(size_t)h * T_LEN * HD + (size_t)t * HD + d] = f2b(ok);
    }
}

// ---------------- V transpose: f32 [t][h*d] -> bf16 [h][d][t] ----------------
__global__ __launch_bounds__(256) void v_transpose_kernel(
    const float* __restrict__ vf, unsigned short* __restrict__ vt)
{
    const int idx = blockIdx.x * 256 + threadIdx.x;
    if (idx >= T_LEN * HID) return;
    const int t = idx >> 11;
    const int c = idx & 2047;
    const int h = c >> 7;
    const int d = c & 127;
    vt[(size_t)h * HD * T_LEN + (size_t)d * T_LEN + t] = f2b(vf[idx]);
}

// ---------------- fused dual-softmax flash attention ----------------
struct __align__(64) WaveLds {
    float          Sf[16 * 32];        // score tile f32
    unsigned short Pg[16 * 32];        // global P (bf16)
    unsigned short Pl[16 * 32];        // local P (bf16)
    float mg[16], lg[16], ag[16];
    float ml[16], ll[16], al[16];
    float gate[16];
};

__global__ __launch_bounds__(256) void attn_kernel(
    const unsigned short* __restrict__ qbf, const unsigned short* __restrict__ kbf,
    const unsigned short* __restrict__ vt,  const float* __restrict__ qg,
    float* __restrict__ attn)
{
    __shared__ WaveLds lds[8];
    const int tid  = threadIdx.x;
    const int lane = tid & 31;
    const int wave = tid >> 5;
    WaveLds& W = lds[wave];

    const int wg = blockIdx.x * 8 + wave;    // 0..2047
    const int h  = wg >> 7;
    const int qb = wg & 127;
    const int t0 = qb * 16;
    const int hi = lane >> 4;
    const int lx = lane & 15;
    const float scale = 0.08838834764831845f;   // 1/sqrt(128)

    // Q A-fragments: 4 chunks over D=128
    Frag aq[4];
    {
        const unsigned short* qrow = qbf + ((size_t)h * T_LEN + (t0 + lx)) * HD;
#pragma unroll
        for (int dc = 0; dc < 4; dc++) {
            aq[dc].h[0] = *(const v8bf*)(qrow + dc * 32 + 8 * hi);
            aq[dc].h[1] = *(const v8bf*)(qrow + dc * 32 + 16 + 8 * hi);
        }
    }
    if (lane < 16) {
        W.mg[lane] = -1e30f; W.lg[lane] = 0.0f;
        W.ml[lane] = -1e30f; W.ll[lane] = 0.0f;
        W.gate[lane] = qg[(size_t)(t0 + lane) * NQG + NH * HD + h];
    }
    lds_fence();

    v8f accg[8], accl[8];
#pragma unroll
    for (int j = 0; j < 8; j++)
#pragma unroll
        for (int r = 0; r < 8; r++) { accg[j][r] = 0.0f; accl[j][r] = 0.0f; }

    const int nkb = (t0 + 15) / 32 + 1;
    for (int kb = 0; kb < nkb; kb++) {
        const int ks = kb * 32;
        const bool has_local = (ks + 31) >= (t0 - (WIN - 1));

        // scores: 2 key sub-tiles x 4 d-chunks of WMMA
#pragma unroll
        for (int sub = 0; sub < 2; sub++) {
            v8f s;
#pragma unroll
            for (int r = 0; r < 8; r++) s[r] = 0.0f;
            const unsigned short* krow = kbf + ((size_t)h * T_LEN + (ks + sub * 16 + lx)) * HD;
            Frag bk;
#pragma unroll
            for (int dc = 0; dc < 4; dc++) {
                bk.h[0] = *(const v8bf*)(krow + dc * 32 + 8 * hi);
                bk.h[1] = *(const v8bf*)(krow + dc * 32 + 16 + 8 * hi);
                s = wmma_bf16(aq[dc].v, bk.v, s);
            }
#pragma unroll
            for (int r = 0; r < 8; r++)
                W.Sf[(r + 8 * hi) * 32 + sub * 16 + lx] = s[r];
        }
        lds_fence();

        // dual online softmax: lanes 0..15 own one query row each
        if (lane < 16) {
            const int t = t0 + lane;
            const float mg_old = W.mg[lane], ml_old = W.ml[lane];
            float rmg = mg_old, rml = ml_old;
#pragma unroll
            for (int jj = 0; jj < 32; jj++) {
                const int key = ks + jj;
                const float x = W.Sf[lane * 32 + jj] * scale;
                const bool cg = (key <= t);
                const bool cl = cg && (t - key < WIN);
                if (cg) rmg = fmaxf(rmg, x);
                if (cl) rml = fmaxf(rml, x);
            }
            float sumg = 0.0f, suml = 0.0f;
#pragma unroll
            for (int jj = 0; jj < 32; jj++) {
                const int key = ks + jj;
                const float x = W.Sf[lane * 32 + jj] * scale;
                const bool cg = (key <= t);
                const bool cl = cg && (t - key < WIN);
                const float pg = cg ? __expf(x - rmg) : 0.0f;
                const float pl = cl ? __expf(x - rml) : 0.0f;
                W.Pg[lane * 32 + jj] = f2b(pg);
                W.Pl[lane * 32 + jj] = f2b(pl);
                sumg += pg; suml += pl;
            }
            const float alg = __expf(mg_old - rmg);
            const float all = __expf(ml_old - rml);
            W.ag[lane] = alg; W.al[lane] = all;
            W.lg[lane] = W.lg[lane] * alg + sumg;
            W.ll[lane] = W.ll[lane] * all + suml;
            W.mg[lane] = rmg; W.ml[lane] = rml;
        }
        lds_fence();

        // rescale accumulators by per-row alpha
        float alg[8], all[8];
#pragma unroll
        for (int r = 0; r < 8; r++) { alg[r] = W.ag[r + 8 * hi]; all[r] = W.al[r + 8 * hi]; }
#pragma unroll
        for (int j = 0; j < 8; j++)
#pragma unroll
            for (int r = 0; r < 8; r++) {
                accg[j][r] *= alg[r];
                if (has_local) accl[j][r] *= all[r];
            }

        // P A-fragments (16x32)
        Frag pga, pla;
        {
            const unsigned short* pg = &W.Pg[lx * 32];
            pga.h[0] = *(const v8bf*)(pg + 8 * hi);
            pga.h[1] = *(const v8bf*)(pg + 16 + 8 * hi);
            const unsigned short* pl = &W.Pl[lx * 32];
            pla.h[0] = *(const v8bf*)(pl + 8 * hi);
            pla.h[1] = *(const v8bf*)(pl + 16 + 8 * hi);
        }
        // PV: 8 d-tiles, B from V^T (contiguous over keys)
#pragma unroll
        for (int j = 0; j < 8; j++) {
            Frag bv;
            const unsigned short* vrow = vt + ((size_t)h * HD + (j * 16 + lx)) * T_LEN + ks + 8 * hi;
            bv.h[0] = *(const v8bf*)(vrow);
            bv.h[1] = *(const v8bf*)(vrow + 16);
            accg[j] = wmma_bf16(pga.v, bv.v, accg[j]);
            if (has_local) accl[j] = wmma_bf16(pla.v, bv.v, accl[j]);
        }
    }

    lds_fence();
    float invg[8], invl[8], gsel[8];
#pragma unroll
    for (int r = 0; r < 8; r++) {
        const int row = r + 8 * hi;
        invg[r] = 1.0f / W.lg[row];
        invl[r] = 1.0f / W.ll[row];
        gsel[r] = W.gate[row];
    }
#pragma unroll
    for (int j = 0; j < 8; j++)
#pragma unroll
        for (int r = 0; r < 8; r++) {
            const int t = t0 + r + 8 * hi;
            const int d = j * 16 + lx;
            const float val = (gsel[r] > 0.0f) ? accg[j][r] * invg[r] : accl[j][r] * invl[r];
            attn[(size_t)t * HID + h * HD + d] = val;
        }
}

// ---------------- launch ----------------
extern "C" void kernel_launch(void* const* d_in, const int* in_sizes, int n_in,
                              void* d_out, int out_size, void* d_ws, size_t ws_size,
                              hipStream_t stream) {
    const float* hidden = (const float*)d_in[0];
    // d_in[1] = positions (arange, folded into RoPE directly)
    const float* Wq  = (const float*)d_in[2];
    const float* Wk  = (const float*)d_in[3];
    const float* Wv  = (const float*)d_in[4];
    const float* Wo  = (const float*)d_in[5];
    const float* qnw = (const float*)d_in[6];
    const float* knw = (const float*)d_in[7];
    float* out = (float*)d_out;

    char* ws = (char*)d_ws;
    size_t off = 0;
    float* qg = (float*)(ws + off);          off += (size_t)T_LEN * NQG * 4;
    float* kf = (float*)(ws + off);          off += (size_t)T_LEN * HID * 4;
    float* vf = (float*)(ws + off);          off += (size_t)T_LEN * HID * 4;
    unsigned short* qbf = (unsigned short*)(ws + off); off += (size_t)NH * T_LEN * HD * 2;
    unsigned short* kbf = (unsigned short*)(ws + off); off += (size_t)NH * T_LEN * HD * 2;
    unsigned short* vtb = (unsigned short*)(ws + off); off += (size_t)NH * HD * T_LEN * 2;
    float* attn = vf;   // vf is dead after v_transpose; alias to save workspace

    dim3 blk(256);
    // QKV projections
    gemm_bf16_kernel<<<dim3((NQG + GBN - 1) / GBN, T_LEN / GBM), blk, 0, stream>>>(
        hidden, Wq, qg, T_LEN, NQG, HID);
    gemm_bf16_kernel<<<dim3(HID / GBN, T_LEN / GBM), blk, 0, stream>>>(
        hidden, Wk, kf, T_LEN, HID, HID);
    gemm_bf16_kernel<<<dim3(HID / GBN, T_LEN / GBM), blk, 0, stream>>>(
        hidden, Wv, vf, T_LEN, HID, HID);
    // norm + rope, V transpose
    norm_rope_kernel<<<T_LEN, blk, 0, stream>>>(qg, kf, qnw, knw, qbf, kbf);
    v_transpose_kernel<<<(T_LEN * HID) / 256, blk, 0, stream>>>(vf, vtb);
    // fused dual attention (one wave per head x 16-query tile)
    attn_kernel<<<(NH * (T_LEN / 16)) / 8, blk, 0, stream>>>(qbf, kbf, vtb, qg, attn);
    // output projection
    gemm_bf16_kernel<<<dim3(HID / GBN, T_LEN / GBM), blk, 0, stream>>>(
        attn, Wo, out, T_LEN, HID, HID);
}
